// EncoderBlock_4647154614531
// MI455X (gfx1250) — compile-verified, matrix operand order checked by reading, not run on hardware
//
#include <hip/hip_runtime.h>
#include <hip/hip_bf16.h>

// ---------------------------------------------------------------------------
// MI455X (gfx1250) transformer encoder block.
// All GEMMs use v_wmma_f32_16x16x32_bf16 (bf16 inputs, fp32 accumulation).
// fp32 operands are converted to bf16 while staging global->LDS.
// LDS tiles are double-buffered: one s_barrier per K-iteration, staging of
// tile k+1 overlaps the 8 WMMAs of tile k.
// ---------------------------------------------------------------------------

typedef __attribute__((ext_vector_type(16))) __bf16          v16bf;
typedef __attribute__((ext_vector_type(8)))  float           v8f;
typedef __attribute__((ext_vector_type(4)))  unsigned short  us4;

#define DEV __device__ __forceinline__

DEV unsigned short f2bf(float f) {
    unsigned int u = __float_as_uint(f);
    u += 0x7fffu + ((u >> 16) & 1u);          // round-to-nearest-even
    return (unsigned short)(u >> 16);
}

// Pack two fp32 -> two bf16 in one dword (v_cvt_pk_bf16_f32 when available).
#if defined(__has_builtin) && __has_builtin(__builtin_amdgcn_cvt_pk_bf16_f32)
DEV unsigned int pk2bf(float a, float b) {
    auto r = __builtin_amdgcn_cvt_pk_bf16_f32(a, b);
    union { decltype(r) v; unsigned int u; } c;
    c.v = r;
    return c.u;
}
#else
DEV unsigned int pk2bf(float a, float b) {
    return (unsigned int)f2bf(a) | ((unsigned int)f2bf(b) << 16);
}
#endif

// K offset of VGPR p (2 packed bf16 per dword) for the 16-bit A/B WMMA layout:
// lanes 0-15 hold K-halves {0..7,16..23}, lanes 16-31 hold {8..15,24..31}.
DEV int kmap(int p, int ksel) {
    return (p < 4) ? (ksel * 8 + 2 * p) : (16 + ksel * 8 + 2 * (p - 4));
}

DEV v16bf load_frag(const unsigned int* lds32, int halfBase, int ksel) {
    union { unsigned int u[8]; v16bf v; } f;
#pragma unroll
    for (int p = 0; p < 8; ++p)
        f.u[p] = lds32[(halfBase + kmap(p, ksel)) >> 1];
    return f.v;
}

DEV float wredmax(float v) {
#pragma unroll
    for (int o = 16; o > 0; o >>= 1) v = fmaxf(v, __shfl_xor(v, o, 32));
    return v;
}
DEV float wredsum(float v) {
#pragma unroll
    for (int o = 16; o > 0; o >>= 1) v += __shfl_xor(v, o, 32);
    return v;
}

// ---------------------------------------------------------------------------
// Batched WMMA GEMM:  C[z] = relu( alpha * A[z] @ B[z] + bias )
//   A: [M x K]  (fp32 or bf16), row-major, lda
//   B: !BTRANS: [K x N] row-major ldb ; BTRANS: [N x K] row-major ldb
//   C: [M x N]  (fp32 or bf16), ldc
//   z = blockIdx.z ; zb = z/innerB, zi = z%innerB; off* = zb*s*o + zi*s*i
// ---------------------------------------------------------------------------
template <int BM, int BN, int MW, int NW,
          bool AF32, bool BF32, bool BTRANS, bool OUTBF, bool RELU>
__global__ __launch_bounds__(256) void gemm_wmma(
    const void* __restrict__ Ap, const void* __restrict__ Bp,
    void* __restrict__ Cp, const float* __restrict__ bias,
    int K, int lda, int ldb, int ldc,
    long sAo, long sAi, long sBo, long sBi, long sCo, long sCi,
    int innerB, float alpha)
{
    constexpr int BK      = 32;
    constexpr int THREADS = 256;
    constexpr int ASTR    = BK + 2;                // padded LDS row (halves)
    constexpr int MT      = BM / (16 * MW);
    constexpr int NT      = BN / (16 * NW);
    constexpr int ASZ     = BM * ASTR;             // halves per A buffer
    constexpr int BSZ     = BN * ASTR;             // halves per B buffer

    __shared__ unsigned short Al[2 * ASZ];         // A tile [m][k], double buffered
    __shared__ unsigned short Bl[2 * BSZ];         // B tile [n][k], double buffered

    const int tid  = threadIdx.x;
    const int lane = tid & 31;
    const int wave = tid >> 5;
    const int wm   = wave % MW;
    const int wn   = wave / MW;

    const int  zb   = blockIdx.z / innerB;
    const int  zi   = blockIdx.z % innerB;
    const long offA = (long)zb * sAo + (long)zi * sAi;
    const long offB = (long)zb * sBo + (long)zi * sBi;
    const long offC = (long)zb * sCo + (long)zi * sCi;

    const int row0 = blockIdx.y * BM;
    const int col0 = blockIdx.x * BN;

    const float*          Af = (const float*)Ap;
    const unsigned short* Ah = (const unsigned short*)Ap;
    const float*          Bf = (const float*)Bp;
    const unsigned short* Bh = (const unsigned short*)Bp;

    v8f acc[MT][NT];
#pragma unroll
    for (int i = 0; i < MT; ++i)
#pragma unroll
        for (int j = 0; j < NT; ++j)
#pragma unroll
            for (int r = 0; r < 8; ++r) acc[i][j][r] = 0.0f;

    // ---- stage one K-tile (A and B) into LDS buffer `buf` ----
    auto stage = [&](int k0, int buf) {
        unsigned int* A32 = (unsigned int*)(Al + buf * ASZ);
        unsigned int* B32 = (unsigned int*)(Bl + buf * BSZ);
        // A: BM x BK, 4 elements along K per thread-iteration
        {
            constexpr int CH = BM * BK / 4;
            for (int c = tid; c < CH; c += THREADS) {
                const int m  = c >> 3;             // BK/4 == 8
                const int k4 = (c & 7) << 2;
                unsigned int u0, u1;
                if (AF32) {
                    const float4 f = *(const float4*)(Af + offA + (long)(row0 + m) * lda + (k0 + k4));
                    u0 = pk2bf(f.x, f.y); u1 = pk2bf(f.z, f.w);
                } else {
                    const us4 f = *(const us4*)(Ah + offA + (long)(row0 + m) * lda + (k0 + k4));
                    u0 = (unsigned int)f[0] | ((unsigned int)f[1] << 16);
                    u1 = (unsigned int)f[2] | ((unsigned int)f[3] << 16);
                }
                const int hb = m * ASTR + k4;      // even
                A32[(hb >> 1) + 0] = u0;
                A32[(hb >> 1) + 1] = u1;
            }
        }
        // B into [n][k] layout
        if (BTRANS) {   // global B is [N x K]: contiguous along K
            constexpr int CH = BN * BK / 4;
            for (int c = tid; c < CH; c += THREADS) {
                const int n  = c >> 3;
                const int k4 = (c & 7) << 2;
                unsigned int u0, u1;
                if (BF32) {
                    const float4 f = *(const float4*)(Bf + offB + (long)(col0 + n) * ldb + (k0 + k4));
                    u0 = pk2bf(f.x, f.y); u1 = pk2bf(f.z, f.w);
                } else {
                    const us4 f = *(const us4*)(Bh + offB + (long)(col0 + n) * ldb + (k0 + k4));
                    u0 = (unsigned int)f[0] | ((unsigned int)f[1] << 16);
                    u1 = (unsigned int)f[2] | ((unsigned int)f[3] << 16);
                }
                const int hb = n * ASTR + k4;
                B32[(hb >> 1) + 0] = u0;
                B32[(hb >> 1) + 1] = u1;
            }
        } else {        // global B is [K x N]: contiguous along N, scatter into [n][k]
            unsigned short* Bh16 = Bl + buf * BSZ;
            constexpr int CH = BK * BN / 4;
            constexpr int NC = BN / 4;
            for (int c = tid; c < CH; c += THREADS) {
                const int k  = c / NC;
                const int n4 = (c % NC) << 2;
                unsigned short h[4];
                if (BF32) {
                    const float4 f = *(const float4*)(Bf + offB + (long)(k0 + k) * ldb + (col0 + n4));
                    h[0] = f2bf(f.x); h[1] = f2bf(f.y); h[2] = f2bf(f.z); h[3] = f2bf(f.w);
                } else {
                    const us4 f = *(const us4*)(Bh + offB + (long)(k0 + k) * ldb + (col0 + n4));
                    h[0] = f[0]; h[1] = f[1]; h[2] = f[2]; h[3] = f[3];
                }
#pragma unroll
                for (int j = 0; j < 4; ++j) Bh16[(n4 + j) * ASTR + k] = h[j];
            }
        }
    };

    const int nk = K / BK;
    stage(0, 0);
    __syncthreads();

    for (int ki = 0; ki < nk; ++ki) {
        const int cur = ki & 1;

        // stage next tile into the other buffer while computing this one
        if (ki + 1 < nk) {
            stage((ki + 1) * BK, cur ^ 1);
            // prefetch the tile after that into L2 (global_prefetch_b8)
            if (ki + 2 < nk) {
                const long e = offA + (long)(row0 + (tid & (BM - 1))) * lda + (ki + 2) * BK;
                __builtin_prefetch(AF32 ? (const void*)(Af + e) : (const void*)(Ah + e), 0, 0);
            }
        }

        const unsigned int* A32 = (const unsigned int*)(Al + cur * ASZ);
        const unsigned int* B32 = (const unsigned int*)(Bl + cur * BSZ);
        const int ksel = lane >> 4;
        const int lr   = lane & 15;
        v16bf aF[MT];
        v16bf bF[NT];
#pragma unroll
        for (int mt = 0; mt < MT; ++mt)
            aF[mt] = load_frag(A32, (wm * MT * 16 + mt * 16 + lr) * ASTR, ksel);
#pragma unroll
        for (int nt = 0; nt < NT; ++nt)
            bF[nt] = load_frag(B32, (wn * NT * 16 + nt * 16 + lr) * ASTR, ksel);

#pragma unroll
        for (int mt = 0; mt < MT; ++mt)
#pragma unroll
            for (int nt = 0; nt < NT; ++nt)
                acc[mt][nt] = __builtin_amdgcn_wmma_f32_16x16x32_bf16(
                    false, aF[mt], false, bF[nt], (short)0, acc[mt][nt], false, false);

        __syncthreads();
    }

    // ---- epilogue: alpha, bias, relu, store (C layout: VGPR r -> row (lane>>4)*8+r, col lane&15) ----
#pragma unroll
    for (int nt = 0; nt < NT; ++nt) {
        const int   c  = col0 + wn * NT * 16 + nt * 16 + (lane & 15);
        const float bv = bias ? bias[c] : 0.0f;
#pragma unroll
        for (int mt = 0; mt < MT; ++mt) {
            const int rbase = row0 + wm * MT * 16 + mt * 16 + ((lane >> 4) * 8);
#pragma unroll
            for (int r = 0; r < 8; ++r) {
                float v = acc[mt][nt][r] * alpha + bv;
                if (RELU) v = fmaxf(v, 0.0f);
                const long idx = offC + (long)(rbase + r) * ldc + c;
                if (OUTBF) ((unsigned short*)Cp)[idx] = f2bf(v);
                else       ((float*)Cp)[idx]          = v;
            }
        }
    }
}

// ---------------------------------------------------------------------------
// In-place masked softmax over rows of 1024 (one wave per row).
// rows = B*H*S = 65536 ; mask is [B,S,S] bool (1 byte each).
// ---------------------------------------------------------------------------
__global__ __launch_bounds__(256) void softmax_rows(
    float* __restrict__ probs, const unsigned char* __restrict__ mask)
{
    const int lane = threadIdx.x & 31;
    const int row  = blockIdx.x * 8 + (threadIdx.x >> 5);
    const int q    = row & 1023;
    const int b    = row >> 14;                   // H*S = 16384 rows per batch

    float4*             pr = (float4*)(probs + (long)row * 1024);
    const unsigned int* mr = (const unsigned int*)(mask + ((long)(b * 1024 + q)) * 1024);

    float4 v[8];
    float  mx = -3.4e38f;
#pragma unroll
    for (int i = 0; i < 8; ++i) {
        const int idx = lane + (i << 5);
        float4 f = pr[idx];
        const unsigned int mm = mr[idx];
        if (mm & 0x000000FFu) f.x = -1e9f;
        if (mm & 0x0000FF00u) f.y = -1e9f;
        if (mm & 0x00FF0000u) f.z = -1e9f;
        if (mm & 0xFF000000u) f.w = -1e9f;
        v[i] = f;
        mx = fmaxf(mx, fmaxf(fmaxf(f.x, f.y), fmaxf(f.z, f.w)));
    }
    mx = wredmax(mx);

    float s = 0.0f;
#pragma unroll
    for (int i = 0; i < 8; ++i) {
        v[i].x = __expf(v[i].x - mx);
        v[i].y = __expf(v[i].y - mx);
        v[i].z = __expf(v[i].z - mx);
        v[i].w = __expf(v[i].w - mx);
        s += v[i].x + v[i].y + v[i].z + v[i].w;
    }
    s = wredsum(s);
    const float inv = 1.0f / s;
#pragma unroll
    for (int i = 0; i < 8; ++i) {
        const int idx = lane + (i << 5);
        float4 f = v[i];
        f.x *= inv; f.y *= inv; f.z *= inv; f.w *= inv;
        pr[idx] = f;
    }
}

// ---------------------------------------------------------------------------
// out = LayerNorm(X + R) * g + be     (one wave per 1024-wide row)
// Writes fp32 always; bf16 copy if outB != nullptr.
// ---------------------------------------------------------------------------
__global__ __launch_bounds__(256) void add_ln(
    const float* __restrict__ X, const float* __restrict__ R,
    const float* __restrict__ g, const float* __restrict__ be,
    float* __restrict__ outF, unsigned short* __restrict__ outB)
{
    const int lane = threadIdx.x & 31;
    const int row  = blockIdx.x * 8 + (threadIdx.x >> 5);

    const float4* x4 = (const float4*)(X + (long)row * 1024);
    const float4* r4 = (const float4*)(R + (long)row * 1024);

    float4 v[8];
    float  s = 0.0f, s2 = 0.0f;
#pragma unroll
    for (int i = 0; i < 8; ++i) {
        const int idx = lane + (i << 5);
        float4 a = x4[idx];
        const float4 b = r4[idx];
        a.x += b.x; a.y += b.y; a.z += b.z; a.w += b.w;
        v[i] = a;
        s  += a.x + a.y + a.z + a.w;
        s2 += a.x * a.x + a.y * a.y + a.z * a.z + a.w * a.w;
    }
    s  = wredsum(s);
    s2 = wredsum(s2);
    const float mu   = s * (1.0f / 1024.0f);
    const float var  = s2 * (1.0f / 1024.0f) - mu * mu;
    const float rstd = rsqrtf(var + 1e-6f);

    const float4* g4 = (const float4*)g;
    const float4* b4 = (const float4*)be;
#pragma unroll
    for (int i = 0; i < 8; ++i) {
        const int idx = lane + (i << 5);
        const float4 gg = g4[idx];
        const float4 bb = b4[idx];
        float4 o;
        o.x = (v[i].x - mu) * rstd * gg.x + bb.x;
        o.y = (v[i].y - mu) * rstd * gg.y + bb.y;
        o.z = (v[i].z - mu) * rstd * gg.z + bb.z;
        o.w = (v[i].w - mu) * rstd * gg.w + bb.w;
        ((float4*)(outF + (long)row * 1024))[idx] = o;
        if (outB) {
            const unsigned int u0 = pk2bf(o.x, o.y);
            const unsigned int u1 = pk2bf(o.z, o.w);
            unsigned int* dst = (unsigned int*)(outB + (long)row * 1024 + idx * 4);
            dst[0] = u0;
            dst[1] = u1;
        }
    }
}

// ---------------------------------------------------------------------------
extern "C" void kernel_launch(void* const* d_in, const int* in_sizes, int n_in,
                              void* d_out, int out_size, void* d_ws, size_t ws_size,
                              hipStream_t stream)
{
    (void)in_sizes; (void)n_in; (void)out_size; (void)ws_size;

    const float*         X    = (const float*)d_in[0];
    const unsigned char* mask = (const unsigned char*)d_in[1];
    const float* Wq = (const float*)d_in[2];  const float* bq  = (const float*)d_in[3];
    const float* Wk = (const float*)d_in[4];  const float* bk  = (const float*)d_in[5];
    const float* Wv = (const float*)d_in[6];  const float* bv  = (const float*)d_in[7];
    const float* Wo = (const float*)d_in[8];  const float* bo  = (const float*)d_in[9];
    const float* W1 = (const float*)d_in[10]; const float* b1  = (const float*)d_in[11];
    const float* W2 = (const float*)d_in[12]; const float* b2  = (const float*)d_in[13];
    const float* g1 = (const float*)d_in[14]; const float* be1 = (const float*)d_in[15];
    const float* g2 = (const float*)d_in[16]; const float* be2 = (const float*)d_in[17];

    const size_t MB = 1024ull * 1024ull;
    char* ws = (char*)d_ws;
    // live ranges allow reuse: h1 overlays Q/K/V/ctx (dead after Wo GEMM),
    // ff overlays attn_out (dead after LN1).  Peak = 72 MB.
    unsigned short* Qb   = (unsigned short*)(ws + 0 * MB);
    unsigned short* Kb   = (unsigned short*)(ws + 8 * MB);
    unsigned short* Vb   = (unsigned short*)(ws + 16 * MB);
    unsigned short* Cx   = (unsigned short*)(ws + 24 * MB);
    float*          attn = (float*)(ws + 32 * MB);
    float*          xln  = (float*)(ws + 48 * MB);
    unsigned short* xlb  = (unsigned short*)(ws + 64 * MB);
    unsigned short* h1   = (unsigned short*)(ws + 0 * MB);   // reuse
    float*          ff   = (float*)(ws + 32 * MB);           // reuse

    float* finalO = (float*)d_out;
    float* probs  = (float*)d_out + 4ull * 1024ull * 1024ull;

    const long M1 = 1024 * 1024;
    const dim3 blk(256);

    // --- QKV projections: [4096x1024] = X @ W + b  -> bf16 ---
    const dim3 gP(1024 / 128, 4096 / 128, 1);
    gemm_wmma<128,128,2,4, true,true,false,true,false><<<gP, blk, 0, stream>>>(
        X, Wq, Qb, bq, 1024, 1024, 1024, 1024, 0,0,0,0,0,0, 1, 1.0f);
    gemm_wmma<128,128,2,4, true,true,false,true,false><<<gP, blk, 0, stream>>>(
        X, Wk, Kb, bk, 1024, 1024, 1024, 1024, 0,0,0,0,0,0, 1, 1.0f);
    gemm_wmma<128,128,2,4, true,true,false,true,false><<<gP, blk, 0, stream>>>(
        X, Wv, Vb, bv, 1024, 1024, 1024, 1024, 0,0,0,0,0,0, 1, 1.0f);

    // --- scores[b,h] = (Q @ K^T) / sqrt(64)  -> fp32 probs region of d_out ---
    const dim3 gS(1024 / 128, 1024 / 128, 64);
    gemm_wmma<128,128,2,4, false,false,true,false,false><<<gS, blk, 0, stream>>>(
        Qb, Kb, probs, nullptr, 64, 1024, 1024, 1024,
        M1, 64, M1, 64, 16 * M1, M1, 16, 0.125f);

    // --- masked softmax in place ---
    softmax_rows<<<65536 / 8, blk, 0, stream>>>(probs, mask);

    // --- context[b,h] = P @ V  -> bf16 [4096, H*DH] ---
    const dim3 gPV(64 / 64, 1024 / 128, 64);
    gemm_wmma<128,64,4,2, true,false,false,true,false><<<gPV, blk, 0, stream>>>(
        probs, Vb, Cx, nullptr, 1024, 1024, 1024, 1024,
        16 * M1, M1, M1, 64, M1, 64, 16, 1.0f);

    // --- attn_out = ctx @ Wo + bo -> fp32 ---
    gemm_wmma<128,128,2,4, false,true,false,false,false><<<gP, blk, 0, stream>>>(
        Cx, Wo, attn, bo, 1024, 1024, 1024, 1024, 0,0,0,0,0,0, 1, 1.0f);

    // --- x = LN(inputs + attn_out) -> fp32 + bf16 ---
    add_ln<<<4096 / 8, blk, 0, stream>>>(X, attn, g1, be1, xln, xlb);

    // --- h1 = relu(x @ W1 + b1) -> bf16 [4096 x 4096] ---
    const dim3 gF1(4096 / 128, 4096 / 128, 1);
    gemm_wmma<128,128,2,4, false,true,false,true,true><<<gF1, blk, 0, stream>>>(
        xlb, W1, h1, b1, 1024, 1024, 4096, 4096, 0,0,0,0,0,0, 1, 1.0f);

    // --- ff = h1 @ W2 + b2 -> fp32 ---
    gemm_wmma<128,128,2,4, false,true,false,false,false><<<gP, blk, 0, stream>>>(
        h1, W2, ff, b2, 4096, 4096, 1024, 1024, 0,0,0,0,0,0, 1, 1.0f);

    // --- final = LN(ff + x) -> d_out ---
    add_ln<<<4096 / 8, blk, 0, stream>>>(ff, xln, g2, be2, finalO, nullptr);
}
